// BiLSTM_CRF_46978352283903
// MI455X (gfx1250) — matrix operand order repaired and checked by reading, not compiled, not run on hardware
//
#include <hip/hip_runtime.h>

#define S_LEN   4096
#define E_DIM   1024
#define HD      512
#define G4      2048      // 4*HD
#define HID     1024
#define T_TAGS  20
#define START_TAG 18
#define STOP_TAG  19
#define NEG_INF  (-10000.0f)

#define NWG     16        // workgroups per LSTM direction
#define UPW     32        // hidden units per workgroup (HD / NWG)

typedef __attribute__((ext_vector_type(16))) __bf16 v16bf;
typedef __attribute__((ext_vector_type(8)))  float  v8f;
typedef __attribute__((ext_vector_type(4)))  unsigned int u32x4;
typedef __attribute__((ext_vector_type(8)))  int i32x8;
typedef __attribute__((ext_vector_type(4)))  int i32x4;

union Frag {
    v16bf v;
    unsigned short u[16];
    uint4 q[2];
};

__device__ __forceinline__ unsigned short f2bf(float f) {
    unsigned int u = __float_as_uint(f);
    unsigned int r = u + 0x7FFFu + ((u >> 16) & 1u);   // RNE to bf16
    return (unsigned short)(r >> 16);
}

__device__ __forceinline__ float sigmoidf(float x) {
    return 1.0f / (1.0f + __expf(-x));
}

// ---------------------------------------------------------------------------
// Kernel 1: embedding gather + f32 -> bf16 cast.  X_bf16[s, e]
// ---------------------------------------------------------------------------
__global__ void embed_cast_kernel(const int* __restrict__ sentence,
                                  const float* __restrict__ embed_table,
                                  unsigned short* __restrict__ X)
{
    int s = blockIdx.x;
    int tok = sentence[s];
    const float* src = embed_table + (size_t)tok * E_DIM;
    unsigned short* dst = X + (size_t)s * E_DIM;
    for (int e = threadIdx.x; e < E_DIM; e += blockDim.x)
        dst[e] = f2bf(src[e]);
}

// ---------------------------------------------------------------------------
// Kernel 2: cast a pair of f32 weight matrices (fwd, bwd) to bf16, stacked.
// ---------------------------------------------------------------------------
__global__ void pair_cast_kernel(const float* __restrict__ a,
                                 const float* __restrict__ b,
                                 unsigned short* __restrict__ W,
                                 long long per_dir)
{
    size_t idx = (size_t)blockIdx.x * blockDim.x + threadIdx.x;
    if (idx >= (size_t)(2 * per_dir)) return;
    float v = (idx < (size_t)per_dir) ? a[idx] : b[idx - per_dir];
    W[idx] = f2bf(v);
}

// ---------------------------------------------------------------------------
// Kernel 3: WMMA GEMM  pre[dir][s][j] = x[s,:] . w_ih[dir][j,:] + b_ih[dir][j]
// Register-blocked: one wave computes a 32x64 output tile:
// 2 A-fragments x 4 B-fragments -> 8 v_wmma_f32_16x16x32_bf16 per K-step.
// ---------------------------------------------------------------------------
__global__ void wmma_pre_gemm(const unsigned short* __restrict__ X,   // [S][E] bf16
                              const unsigned short* __restrict__ W,   // [2][G4][E] bf16
                              const float* __restrict__ b_ih_f,
                              const float* __restrict__ b_ih_b,
                              float* __restrict__ pre)                // [2][S][G4]
{
    int bid  = blockIdx.x;
    int dir  = bid >> 12;              // 2 dirs
    int rem  = bid & 4095;             // 128 * 32 tiles
    int tn4  = rem & 31;               // 64-wide N tile
    int tm2  = rem >> 5;               // 32-tall M tile
    int lane = threadIdx.x;            // wave32
    int hl   = lane >> 4;              // half-wave: 0 | 1
    int l15  = lane & 15;

    const unsigned short* Wd = W + (size_t)dir * G4 * E_DIM;
    int m_base = tm2 * 32;
    int n_base = tn4 * 64;

    const unsigned short* Arow0 = X + (size_t)(m_base + l15)      * E_DIM + (hl ? 8 : 0);
    const unsigned short* Arow1 = X + (size_t)(m_base + 16 + l15) * E_DIM + (hl ? 8 : 0);
    const unsigned short* Brow[4];
#pragma unroll
    for (int nn = 0; nn < 4; ++nn)
        Brow[nn] = Wd + (size_t)(n_base + nn * 16 + l15) * E_DIM + (hl ? 8 : 0);

    v8f c[2][4];
#pragma unroll
    for (int mi = 0; mi < 2; ++mi)
#pragma unroll
        for (int nn = 0; nn < 4; ++nn)
            c[mi][nn] = (v8f){};

    for (int k0 = 0; k0 < E_DIM; k0 += 32) {
        Frag a[2], b[4];
        a[0].q[0] = *(const uint4*)(Arow0 + k0);
        a[0].q[1] = *(const uint4*)(Arow0 + k0 + 16);
        a[1].q[0] = *(const uint4*)(Arow1 + k0);
        a[1].q[1] = *(const uint4*)(Arow1 + k0 + 16);
#pragma unroll
        for (int nn = 0; nn < 4; ++nn) {
            b[nn].q[0] = *(const uint4*)(Brow[nn] + k0);
            b[nn].q[1] = *(const uint4*)(Brow[nn] + k0 + 16);
        }
#pragma unroll
        for (int mi = 0; mi < 2; ++mi)
#pragma unroll
            for (int nn = 0; nn < 4; ++nn)
                c[mi][nn] = __builtin_amdgcn_wmma_f32_16x16x32_bf16(
                        false, a[mi].v, false, b[nn].v,
                        (short)0, c[mi][nn], false, false);
    }

    const float* bih = dir ? b_ih_b : b_ih_f;
    float* out = pre + (size_t)dir * S_LEN * G4;
#pragma unroll
    for (int nn = 0; nn < 4; ++nn) {
        int n = n_base + nn * 16 + l15;
        float bias = bih[n];
#pragma unroll
        for (int mi = 0; mi < 2; ++mi) {
#pragma unroll
            for (int r = 0; r < 8; ++r) {
                int m = m_base + mi * 16 + r + (hl ? 8 : 0);
                out[(size_t)m * G4 + n] = c[mi][nn][r] + bias;
            }
        }
    }
}

// ---------------------------------------------------------------------------
// Kernel 4: init device-scope barrier counters and h double-buffer.
// ---------------------------------------------------------------------------
__global__ void init_state_kernel(const float* __restrict__ h0,
                                  float* __restrict__ hbuf,       // [2][2][HD]
                                  unsigned int* __restrict__ ctr) // [2]
{
    int t = threadIdx.x + blockIdx.x * blockDim.x;
    if (t < 2) ctr[t] = 0;
    if (t < 2 * HD) {
        int dir = t / HD, u = t % HD;
        hbuf[(size_t)dir * 2 * HD + u] = h0[t];   // parity-0 buffer = h0
    }
}

// ---------------------------------------------------------------------------
// Kernel 5: multi-workgroup LSTM recurrence (see header comment in analysis).
// ---------------------------------------------------------------------------
__global__ void lstm_multi_kernel(const float* __restrict__ preBase,   // [2][S][G4]
                                  const unsigned short* __restrict__ WhhBf, // [2][G4][HD]
                                  const float* __restrict__ b_hh_f,
                                  const float* __restrict__ b_hh_b,
                                  const float* __restrict__ c0,
                                  float* __restrict__ hbuf,            // [2][2][HD]
                                  unsigned int* __restrict__ ctr,      // [2]
                                  float* __restrict__ lstm_out)        // [S][HID]
{
    int bid = blockIdx.x;
    int dir = bid >> 4;
    int g   = bid & (NWG - 1);
    int u0  = g * UPW;
    int t   = threadIdx.x;       // 0..255

    const float* pre = preBase + (size_t)dir * S_LEN * G4;
    const unsigned short* whh = WhhBf + (size_t)dir * G4 * HD;
    const float* bhh = dir ? b_hh_b : b_hh_f;
    float* hg = hbuf + (size_t)dir * 2 * HD;
    unsigned int* cntr = ctr + dir;

    __shared__ __align__(16) unsigned int Wl[4 * UPW * (HD / 2)];  // 128 KB bf16 pairs
    __shared__ __align__(16) float h_l[HD];
    __shared__ float part[256];
    __shared__ float gd[4 * UPW];
    __shared__ float c_l[UPW];

#if __has_builtin(__builtin_amdgcn_tensor_load_to_lds)
    // TDM staging: 4 tiles of (32 rows x 512 bf16), row stride 512 elements.
    if (t < 32) {
        unsigned long long gbase = (unsigned long long)(uintptr_t)whh;
        unsigned int ldsbase = (unsigned int)(uintptr_t)(&Wl[0]);
        for (int gi = 0; gi < 4; ++gi) {
            unsigned long long ga = gbase +
                (unsigned long long)(gi * HD + u0) * HD * 2ull;
            unsigned int lds = ldsbase + (unsigned int)(gi * UPW * (HD / 2) * 4);
            u32x4 g0;
            g0[0] = 1u;                                        // count=1, user mode
            g0[1] = lds;                                       // lds_addr
            g0[2] = (unsigned int)(ga & 0xFFFFFFFFu);          // global_addr lo
            g0[3] = (unsigned int)((ga >> 32) & 0x01FFFFFFu)   // global_addr hi
                    | (2u << 30);                              // D# type = 2
            i32x8 g1;
            g1[0] = (int)(1u << 16);                  // data_size = 1 (2 bytes)
            g1[1] = (int)((HD & 0xFFFFu) << 16);      // tensor_dim0 = 512 (lo16)
            g1[2] = (int)((UPW & 0xFFFFu) << 16);     // tensor_dim1 = 32 (lo16)
            g1[3] = (int)((HD & 0xFFFFu) << 16);      // tile_dim0 = 512
            g1[4] = (int)(UPW);                       // tile_dim1 = 32
            g1[5] = (int)(HD);                        // tensor_dim0_stride = 512
            g1[6] = 0;
            g1[7] = 0;
            i32x4 z4 = {0, 0, 0, 0};
#if __clang_major__ >= 23
            i32x8 z8 = {0, 0, 0, 0, 0, 0, 0, 0};
            __builtin_amdgcn_tensor_load_to_lds(g0, g1, z4, z4, z8, 0);
#else
            __builtin_amdgcn_tensor_load_to_lds(g0, g1, z4, z4, 0);
#endif
        }
#if __has_builtin(__builtin_amdgcn_s_wait_tensorcnt)
        __builtin_amdgcn_s_wait_tensorcnt(0);
#endif
    }
#endif

    // Redundant generic-load pass (same values) — correctness does not depend
    // on the TDM descriptor bit packing.
    {
        const unsigned int* wsrc = (const unsigned int*)whh;   // [G4][HD/2] pairs
        for (int i = t; i < 4 * UPW * (HD / 2); i += 256) {
            int r  = i >> 8;            // local row 0..127
            int k  = i & 255;
            int gi = r >> 5, ul = r & 31;
            int j  = gi * HD + u0 + ul;
            Wl[i] = wsrc[(size_t)j * (HD / 2) + k];
        }
    }
    if (t < UPW) c_l[t] = c0[dir * HD + u0 + t];
    __syncthreads();

    unsigned int target = 0;
    for (int step = 0; step < S_LEN; ++step) {
        int s  = dir ? (S_LEN - 1 - step) : step;
        int sn = dir ? (s > 0 ? s - 1 : 0) : (s < S_LEN - 1 ? s + 1 : s);
        const float* hsrc = hg + (size_t)(step & 1) * HD;

        for (int i = t; i < HD; i += 256) h_l[i] = hsrc[i];
        __builtin_prefetch(pre + (size_t)sn * G4 + t * 8, 0, 0);
        __syncthreads();

        {
            int r     = t >> 1;
            int halfk = t & 1;
            const unsigned int* wr = &Wl[r * 256 + halfk * 128];
            const float* hh = &h_l[halfk * 256];
            float acc = 0.0f;
#pragma unroll 4
            for (int k = 0; k < 128; k += 4) {
                uint4  w  = *(const uint4*)(wr + k);
                float4 ha = *(const float4*)(hh + 2 * k);
                float4 hb = *(const float4*)(hh + 2 * k + 4);
                acc = fmaf(__uint_as_float(w.x << 16),         ha.x, acc);
                acc = fmaf(__uint_as_float(w.x & 0xFFFF0000u), ha.y, acc);
                acc = fmaf(__uint_as_float(w.y << 16),         ha.z, acc);
                acc = fmaf(__uint_as_float(w.y & 0xFFFF0000u), ha.w, acc);
                acc = fmaf(__uint_as_float(w.z << 16),         hb.x, acc);
                acc = fmaf(__uint_as_float(w.z & 0xFFFF0000u), hb.y, acc);
                acc = fmaf(__uint_as_float(w.w << 16),         hb.z, acc);
                acc = fmaf(__uint_as_float(w.w & 0xFFFF0000u), hb.w, acc);
            }
            part[t] = acc;
        }
        __syncthreads();

        if (t < 4 * UPW) {
            int gi = t >> 5, ul = t & 31;
            int j  = gi * HD + u0 + ul;
            gd[t]  = part[2 * t] + part[2 * t + 1]
                   + pre[(size_t)s * G4 + j] + bhh[j];
        }
        __syncthreads();

        if (t < UPW) {
            float ig = sigmoidf(gd[t]);
            float fg = sigmoidf(gd[UPW + t]);
            float gg = tanhf(gd[2 * UPW + t]);
            float og = sigmoidf(gd[3 * UPW + t]);
            float cN = fg * c_l[t] + ig * gg;
            c_l[t] = cN;
            float h = og * tanhf(cN);
            lstm_out[(size_t)s * HID + dir * HD + u0 + t] = h;
            hg[(size_t)((step + 1) & 1) * HD + u0 + t] = h;   // other parity
        }

        __threadfence();
        __syncthreads();
        target += NWG;
        if (t == 0) {
            __hip_atomic_fetch_add(cntr, 1u, __ATOMIC_ACQ_REL,
                                   __HIP_MEMORY_SCOPE_AGENT);
#if __has_builtin(__builtin_amdgcn_s_cluster_barrier)
            __builtin_amdgcn_s_cluster_barrier();  // NOP outside a cluster
#endif
            while (__hip_atomic_load(cntr, __ATOMIC_ACQUIRE,
                                     __HIP_MEMORY_SCOPE_AGENT) < target)
                __builtin_amdgcn_s_sleep(1);
        }
        __syncthreads();
    }
}

// ---------------------------------------------------------------------------
// Kernel 6: output projection  feats[s][t] = lstm_out[s] . w_out[t] + b_out[t]
// ---------------------------------------------------------------------------
__global__ void out_proj_kernel(const float* __restrict__ H,
                                const float* __restrict__ Wo,
                                const float* __restrict__ bo,
                                float* __restrict__ feats)
{
    int s = blockIdx.x;
    int t = threadIdx.x;
    if (t >= T_TAGS) return;
    const float4* h4 = (const float4*)(H  + (size_t)s * HID);
    const float4* w4 = (const float4*)(Wo + (size_t)t * HID);
    float acc = bo[t];
    for (int k = 0; k < HID / 4; ++k) {
        float4 a = h4[k];
        float4 b = w4[k];
        acc = fmaf(a.x, b.x, fmaf(a.y, b.y, fmaf(a.z, b.z, fmaf(a.w, b.w, acc))));
    }
    feats[(size_t)s * T_TAGS + t] = acc;
}

// ---------------------------------------------------------------------------
// Kernel 7: Viterbi decode (single block), writes score + int path to d_out.
// ---------------------------------------------------------------------------
__global__ void viterbi_kernel(const float* __restrict__ feats,
                               const float* __restrict__ trans,
                               int* __restrict__ bptr,
                               float* __restrict__ outScore,
                               int* __restrict__ outPath)
{
    __shared__ float fv[T_TAGS];
    __shared__ float fvn[T_TAGS];
    int t = threadIdx.x;
    if (t < T_TAGS) fv[t] = (t == START_TAG) ? 0.0f : NEG_INF;
    __syncthreads();

    for (int s = 0; s < S_LEN; ++s) {
        if (t < T_TAGS) {
            float best = -3.4e38f;
            int bi = 0;
#pragma unroll
            for (int p = 0; p < T_TAGS; ++p) {
                float v = fv[p] + trans[t * T_TAGS + p];
                if (v > best) { best = v; bi = p; }
            }
            bptr[(size_t)s * T_TAGS + t] = bi;
            fvn[t] = best + feats[(size_t)s * T_TAGS + t];
        }
        __syncthreads();
        if (t < T_TAGS) fv[t] = fvn[t];
        __syncthreads();
    }

    if (t == 0) {
        float best = -3.4e38f;
        int bi = 0;
        for (int p = 0; p < T_TAGS; ++p) {
            float v = fv[p] + trans[STOP_TAG * T_TAGS + p];
            if (v > best) { best = v; bi = p; }
        }
        *outScore = best;
        int tag = bi;
        for (int s = S_LEN - 1; s >= 0; --s) {
            outPath[s] = tag;
            tag = bptr[(size_t)s * T_TAGS + tag];
        }
    }
}

// ---------------------------------------------------------------------------
extern "C" void kernel_launch(void* const* d_in, const int* in_sizes, int n_in,
                              void* d_out, int out_size, void* d_ws, size_t ws_size,
                              hipStream_t stream) {
    const int*   sentence    = (const int*)  d_in[0];
    const float* embed_table = (const float*)d_in[1];
    const float* w_ih_f      = (const float*)d_in[2];
    const float* w_hh_f      = (const float*)d_in[3];
    const float* b_ih_f      = (const float*)d_in[4];
    const float* b_hh_f      = (const float*)d_in[5];
    const float* w_ih_b      = (const float*)d_in[6];
    const float* w_hh_b      = (const float*)d_in[7];
    const float* b_ih_b      = (const float*)d_in[8];
    const float* b_hh_b      = (const float*)d_in[9];
    const float* w_out       = (const float*)d_in[10];
    const float* b_out       = (const float*)d_in[11];
    const float* trans       = (const float*)d_in[12];
    const float* h0          = (const float*)d_in[13];
    const float* c0          = (const float*)d_in[14];

    char* ws = (char*)d_ws;
    const size_t MB = 1024 * 1024;
    unsigned short* X    = (unsigned short*)(ws);              // 8 MB
    unsigned short* Wih  = (unsigned short*)(ws + 8  * MB);    // 8 MB
    float* pre           = (float*)(ws + 16 * MB);             // 64 MB
    float* lstm_out      = (float*)(ws + 80 * MB);             // 16 MB
    float* feats         = (float*)(ws + 96 * MB);             // 320 KB
    int*   bptr          = (int*)  (ws + 96 * MB + (size_t)S_LEN * T_TAGS * 4);
    float* hbuf          = (float*)(ws + 97 * MB);             // 8 KB
    unsigned int* ctr    = (unsigned int*)(ws + 97 * MB + 16384);
    unsigned short* Whh  = (unsigned short*)(ws + 98 * MB);    // 4 MB

    embed_cast_kernel<<<S_LEN, 256, 0, stream>>>(sentence, embed_table, X);

    {
        long long per_ih = (long long)G4 * E_DIM;
        int grd = (int)((2 * per_ih + 255) / 256);
        pair_cast_kernel<<<grd, 256, 0, stream>>>(w_ih_f, w_ih_b, Wih, per_ih);
        long long per_hh = (long long)G4 * HD;
        grd = (int)((2 * per_hh + 255) / 256);
        pair_cast_kernel<<<grd, 256, 0, stream>>>(w_hh_f, w_hh_b, Whh, per_hh);
    }

    wmma_pre_gemm<<<2 * (S_LEN / 32) * (G4 / 64), 32, 0, stream>>>(
        X, Wih, b_ih_f, b_ih_b, pre);

    init_state_kernel<<<2, 1024, 0, stream>>>(h0, hbuf, ctr);

    lstm_multi_kernel<<<2 * NWG, 256, 0, stream>>>(pre, Whh, b_hh_f, b_hh_b,
                                                   c0, hbuf, ctr, lstm_out);

    out_proj_kernel<<<S_LEN, 32, 0, stream>>>(lstm_out, w_out, b_out, feats);

    viterbi_kernel<<<1, 32, 0, stream>>>(feats, trans, bptr,
                                         (float*)d_out, (int*)d_out + 1);
}
